// GraphConv_11338713661758
// MI455X (gfx1250) — matrix operand order repaired and checked by reading, not compiled, not run on hardware
//
#include <hip/hip_runtime.h>
#include <hip/hip_bf16.h>

// ---------------------------------------------------------------------------
// GNN step for bipartite graph (NeuroSAT-like), D = 128.
//   4 message MLPs (128->128->128, relu/relu) over full node tables (WMMA bf16)
//   4 edge scatter-adds (atomic f32)
//   2 update MLPs (concat 256->128->128) fused, no concat buffer (WMMA bf16)
// ---------------------------------------------------------------------------

typedef __bf16 bf16_t;
typedef __attribute__((ext_vector_type(16))) __bf16 bf16x16;
typedef __attribute__((ext_vector_type(8)))  float  fx8;

#define HD 128

// Build a 16x32 bf16 A-fragment from fp32 row data.
// Contiguous layout: lanes 0-15 -> K {0..15}; lanes 16-31 -> K {16..31}.
// Caller passes p already offset by (lane>=16 ? 16 : 0) elements.
__device__ __forceinline__ bf16x16 a_frag_from_f32(const float* __restrict__ p) {
    bf16x16 a;
#pragma unroll
    for (int i = 0; i < 8; ++i) {
        a[i]     = (__bf16)p[i];
        a[i + 8] = (__bf16)p[8 + i];
    }
    return a;
}

// Fused 2-layer MLP: Y = relu( relu(X @ W1 + b1) @ W2 + b2 )
//   X  : N x K1 fp32, given as Xa (cols 0..127) and optionally Xb (cols 128..255)
//   W1t: 128 x K1 bf16 (transposed: column n of W1 contiguous in k)
//   W2t: 128 x 128 bf16 (transposed)
//   Y  : N x 128 fp32
// Block: 256 threads = 8 waves; each wave computes a 16-row strip, all 128 cols.
__global__ __launch_bounds__(256)
void mlp2_wmma_kernel(const float* __restrict__ Xa, const float* __restrict__ Xb,
                      const bf16_t* __restrict__ W1t, const float* __restrict__ b1,
                      const bf16_t* __restrict__ W2t, const float* __restrict__ b2,
                      float* __restrict__ Y, int N, int K1)
{
    __shared__ bf16_t Hlds[8][16][HD + 8];   // +8 bf16 pad: row stride 272B kills bank conflicts

    const int wave = threadIdx.x >> 5;
    const int lane = threadIdx.x & 31;
    const int hl   = lane & 15;              // row-within-strip (A) / column (B,C)
    const int hi16 = lane >> 4;              // 0: lanes 0-15, 1: lanes 16-31
    const int koff = hi16 ? 16 : 0;
    const int rowbase = blockIdx.x * 128 + wave * 16;

    // ---------------- layer 1: H = relu(X @ W1 + b1) ----------------
    fx8 acc[8];
#pragma unroll
    for (int nt = 0; nt < 8; ++nt) {
        const float bv = b1[nt * 16 + hl];
#pragma unroll
        for (int v = 0; v < 8; ++v) acc[nt][v] = bv;
    }

    const int row  = rowbase + hl;
    const int lrow = (row < N) ? row : 0;    // clamp loads; keep EXEC all-ones for WMMA

    const int nkb = K1 >> 5;
    for (int kb = 0; kb < nkb; ++kb) {
        const int kg = kb * 32;
        const float* src = (kg < HD) ? (Xa + (size_t)lrow * HD + kg)
                                     : (Xb + (size_t)lrow * HD + (kg - HD));
        const bf16x16 afrag = a_frag_from_f32(src + koff);
        // B fragment: lane holds column (nt*16+hl), K kg+{0..15 | 16..31} contiguous (32B)
        const bf16_t* wcol = W1t + kg + (hi16 ? 16 : 0) + (size_t)hl * K1;
#pragma unroll
        for (int nt = 0; nt < 8; ++nt) {
            const bf16x16 bfrag = *(const bf16x16*)(wcol + (size_t)(nt * 16) * K1);
            acc[nt] = __builtin_amdgcn_wmma_f32_16x16x32_bf16(
                false, afrag, false, bfrag, (short)0, acc[nt], false, false);
        }
    }

    // relu + stash hidden strip as bf16 in per-wave LDS (row-major)
#pragma unroll
    for (int nt = 0; nt < 8; ++nt) {
        const int col = nt * 16 + hl;
#pragma unroll
        for (int v = 0; v < 8; ++v) {
            float h = acc[nt][v];
            h = h > 0.f ? h : 0.f;
            Hlds[wave][v + (hi16 ? 8 : 0)][col] = (__bf16)h;
        }
    }
    __syncthreads();

    // ---------------- layer 2: Y = relu(H @ W2 + b2), K = 128 ----------------
    fx8 acc2[8];
#pragma unroll
    for (int nt = 0; nt < 8; ++nt) {
        const float bv = b2[nt * 16 + hl];
#pragma unroll
        for (int v = 0; v < 8; ++v) acc2[nt][v] = bv;
    }

#pragma unroll
    for (int kb = 0; kb < 4; ++kb) {
        const bf16_t* hp = &Hlds[wave][hl][kb * 32 + koff];
        bf16x16 afrag;
#pragma unroll
        for (int i = 0; i < 8; ++i) { afrag[i] = hp[i]; afrag[i + 8] = hp[8 + i]; }
        const bf16_t* wcol = W2t + kb * 32 + (hi16 ? 16 : 0) + (size_t)hl * HD;
#pragma unroll
        for (int nt = 0; nt < 8; ++nt) {
            const bf16x16 bfrag = *(const bf16x16*)(wcol + (size_t)(nt * 16) * HD);
            acc2[nt] = __builtin_amdgcn_wmma_f32_16x16x32_bf16(
                false, afrag, false, bfrag, (short)0, acc2[nt], false, false);
        }
    }

    // relu + store (predicated per-row; all WMMA already done)
#pragma unroll
    for (int nt = 0; nt < 8; ++nt) {
        const int col = nt * 16 + hl;
#pragma unroll
        for (int v = 0; v < 8; ++v) {
            const int r = rowbase + v + (hi16 ? 8 : 0);
            if (r < N) {
                float y = acc2[nt][v];
                y = y > 0.f ? y : 0.f;
                Y[(size_t)r * HD + col] = y;
            }
        }
    }
}

// acc[dst[e]] += T[src[e]] over 128 floats; 32 lanes per edge, float4 per lane.
__global__ __launch_bounds__(256)
void scatter_add_kernel(const float* __restrict__ T, const int* __restrict__ src,
                        const int* __restrict__ dst, float* __restrict__ acc, int nE)
{
    const size_t t = (size_t)blockIdx.x * blockDim.x + threadIdx.x;
    const int e = (int)(t >> 5);
    if (e >= nE) return;
    const int c = ((int)t & 31) << 2;
    const float4 s = *(const float4*)(T + (size_t)src[e] * HD + c);
    float* d = acc + (size_t)dst[e] * HD + c;
    atomicAdd(d + 0, s.x);
    atomicAdd(d + 1, s.y);
    atomicAdd(d + 2, s.z);
    atomicAdd(d + 3, s.w);
}

// Transpose + convert weights: W (K x 128 fp32, row-major) -> Wt (128 x K bf16)
__global__ __launch_bounds__(256)
void wconv_kernel(const float* __restrict__ W, bf16_t* __restrict__ Wt, int K)
{
    const int t = blockIdx.x * blockDim.x + threadIdx.x;
    if (t >= K * HD) return;
    const int k = t >> 7;
    const int n = t & 127;
    Wt[(size_t)n * K + k] = (__bf16)W[(size_t)k * HD + n];
}

__global__ __launch_bounds__(256)
void fill_zero_kernel(float* __restrict__ p, size_t n)
{
    const size_t t = (size_t)blockIdx.x * blockDim.x + threadIdx.x;
    if (t < n) p[t] = 0.f;
}

extern "C" void kernel_launch(void* const* d_in, const int* in_sizes, int n_in,
                              void* d_out, int out_size, void* d_ws, size_t ws_size,
                              hipStream_t stream)
{
    (void)n_in; (void)out_size; (void)ws_size;

    const float* hv = (const float*)d_in[0];
    const float* hc = (const float*)d_in[1];

    // MLP order: fmv_pos, fmv_neg, fmc_pos, fmc_neg, fuv, fuc
    const float* W1[6]; const float* B1[6]; const float* W2[6]; const float* B2[6];
    for (int i = 0; i < 6; ++i) {
        W1[i] = (const float*)d_in[2 + 4 * i + 0];
        B1[i] = (const float*)d_in[2 + 4 * i + 1];
        W2[i] = (const float*)d_in[2 + 4 * i + 2];
        B2[i] = (const float*)d_in[2 + 4 * i + 3];
    }
    const int* pos_v = (const int*)d_in[26];
    const int* pos_c = (const int*)d_in[27];
    const int* neg_v = (const int*)d_in[28];
    const int* neg_c = (const int*)d_in[29];

    const int NV = in_sizes[0] / HD;
    const int NC = in_sizes[1] / HD;
    const int EP = in_sizes[26];
    const int EN = in_sizes[28];
    const int NMAX = (NC > NV) ? NC : NV;

    // ---- workspace carve-up ----
    char* ws = (char*)d_ws;
    float* T  = (float*)ws;  ws += (size_t)NMAX * HD * sizeof(float);   // reusable MLP table
    float* mv = (float*)ws;  ws += (size_t)NV   * HD * sizeof(float);   // messages to vars
    float* mc = (float*)ws;  ws += (size_t)NC   * HD * sizeof(float);   // messages to clauses
    const int K1s[6] = {128, 128, 128, 128, 256, 256};
    bf16_t* W1t[6]; bf16_t* W2t[6];
    {
        bf16_t* wp = (bf16_t*)ws;
        for (int i = 0; i < 6; ++i) {
            W1t[i] = wp; wp += (size_t)K1s[i] * HD;
            W2t[i] = wp; wp += (size_t)HD * HD;
        }
    }

    // ---- convert/transpose weights to bf16 (tiny) ----
    for (int i = 0; i < 6; ++i) {
        const int n1 = K1s[i] * HD;
        wconv_kernel<<<(n1 + 255) / 256, 256, 0, stream>>>(W1[i], W1t[i], K1s[i]);
        wconv_kernel<<<(HD * HD + 255) / 256, 256, 0, stream>>>(W2[i], W2t[i], HD);
    }

    // ---- zero message accumulators ----
    {
        const size_t nv = (size_t)NV * HD, nc = (size_t)NC * HD;
        fill_zero_kernel<<<(unsigned)((nv + 255) / 256), 256, 0, stream>>>(mv, nv);
        fill_zero_kernel<<<(unsigned)((nc + 255) / 256), 256, 0, stream>>>(mc, nc);
    }

    const unsigned gb_v = (unsigned)((NV + 127) / 128);
    const unsigned gb_c = (unsigned)((NC + 127) / 128);
    const unsigned gs_p = (unsigned)(((size_t)EP * 32 + 255) / 256);
    const unsigned gs_n = (unsigned)(((size_t)EN * 32 + 255) / 256);

    // ---- message phase: mv += seg(fmv_pos(hc))[pos] + seg(fmv_neg(hc))[neg] ----
    mlp2_wmma_kernel<<<gb_c, 256, 0, stream>>>(hc, nullptr, W1t[0], B1[0], W2t[0], B2[0], T, NC, 128);
    scatter_add_kernel<<<gs_p, 256, 0, stream>>>(T, pos_c, pos_v, mv, EP);
    mlp2_wmma_kernel<<<gb_c, 256, 0, stream>>>(hc, nullptr, W1t[1], B1[1], W2t[1], B2[1], T, NC, 128);
    scatter_add_kernel<<<gs_n, 256, 0, stream>>>(T, neg_c, neg_v, mv, EN);

    // ---- message phase: mc += seg(fmc_pos(hv))[pos] + seg(fmc_neg(hv))[neg] ----
    mlp2_wmma_kernel<<<gb_v, 256, 0, stream>>>(hv, nullptr, W1t[2], B1[2], W2t[2], B2[2], T, NV, 128);
    scatter_add_kernel<<<gs_p, 256, 0, stream>>>(T, pos_v, pos_c, mc, EP);
    mlp2_wmma_kernel<<<gb_v, 256, 0, stream>>>(hv, nullptr, W1t[3], B1[3], W2t[3], B2[3], T, NV, 128);
    scatter_add_kernel<<<gs_n, 256, 0, stream>>>(T, neg_v, neg_c, mc, EN);

    // ---- update phase: fused concat (K=256), outputs straight to d_out ----
    float* hv_out = (float*)d_out;
    float* hc_out = (float*)d_out + (size_t)NV * HD;
    mlp2_wmma_kernel<<<gb_v, 256, 0, stream>>>(hv, mv, W1t[4], B1[4], W2t[4], B2[4], hv_out, NV, 256);
    mlp2_wmma_kernel<<<gb_c, 256, 0, stream>>>(hc, mc, W1t[5], B1[5], W2t[5], B2[5], hc_out, NC, 256);
}
